// TFAttention_71906342469919
// MI455X (gfx1250) — matrix operand order repaired and checked by reading, not compiled, not run on hardware
//
#include <hip/hip_runtime.h>
#include <cmath>

typedef __attribute__((ext_vector_type(16))) _Float16 v16h;
typedef __attribute__((ext_vector_type(8)))  _Float16 v8h;
typedef __attribute__((ext_vector_type(8)))  float    v8f;

#define B_   4
#define C_   64
#define T_   1024
#define F_   65
#define H_   4
#define HD_  8
#define CVH_ 16
#define DQK  520      // HD*F
#define DQKP 544      // padded to 17*32: no K-loop tail
#define DV   1040     // CVH*F
#define DVP  1088     // padded to 17*64: 17 macro-columns of 64
#define NH   16       // H*B

// ---------------------------------------------------------------------------
// Stage 1: fused QKV 1x1 conv + PReLU + cfLN, emit f16 Q (pre-scaled by the
// softmax scale), f16 K (row-major, d padded to 544) and V transposed
// (n, DVP, T) so PV B-fragments are contiguous.  One block per (t, b).
// ---------------------------------------------------------------------------
__global__ __launch_bounds__(256) void qkv_proj_kernel(
    const float* __restrict__ x,
    const float* __restrict__ qw, const float* __restrict__ qb, const float* __restrict__ qa,
    const float* __restrict__ qg, const float* __restrict__ qbe,
    const float* __restrict__ kw, const float* __restrict__ kb, const float* __restrict__ ka,
    const float* __restrict__ kg, const float* __restrict__ kbe,
    const float* __restrict__ vw, const float* __restrict__ vb, const float* __restrict__ va,
    const float* __restrict__ vg, const float* __restrict__ vbe,
    _Float16* __restrict__ Qh, _Float16* __restrict__ Kh, _Float16* __restrict__ Vt,
    float scale)
{
  __shared__ float xs[C_ * F_];            // x[b,:,t,:]
  __shared__ float ys[128 * F_];           // 32 Q rows, 32 K rows, 64 V rows
  __shared__ float wqs[H_ * HD_ * C_];
  __shared__ float wks[H_ * HD_ * C_];
  __shared__ float wvs[H_ * CVH_ * C_];
  __shared__ float bqs[32], bks[32], bvs[64];
  __shared__ float segsum[12], segss[12], smu[12], srstd[12];

  const int tid = threadIdx.x;
  const int t = blockIdx.x, b = blockIdx.y;

  for (int i = tid; i < C_ * F_; i += 256) {
    int c = i / F_, f = i % F_;
    xs[i] = x[(((size_t)b * C_ + c) * T_ + t) * F_ + f];
  }
  for (int i = tid; i < H_ * HD_ * C_; i += 256) { wqs[i] = qw[i]; wks[i] = kw[i]; }
  for (int i = tid; i < H_ * CVH_ * C_; i += 256) wvs[i] = vw[i];
  if (tid < 32) { bqs[tid] = qb[tid]; bks[tid] = kb[tid]; }
  if (tid < 64) bvs[tid] = vb[tid];
  if (tid < 12) { segsum[tid] = 0.f; segss[tid] = 0.f; }
  __syncthreads();

  for (int idx = tid; idx < 128 * F_; idx += 256) {
    int r = idx / F_, f = idx % F_;
    const float* wrow; float bias, alpha; int seg;
    if (r < 32)      { int h = r >> 3;                   wrow = &wqs[r * C_];   bias = bqs[r];  alpha = qa[h]; seg = h;     }
    else if (r < 64) { int rr = r - 32; int h = rr >> 3; wrow = &wks[rr * C_];  bias = bks[rr]; alpha = ka[h]; seg = 4 + h; }
    else             { int rr = r - 64; int h = rr >> 4; wrow = &wvs[rr * C_];  bias = bvs[rr]; alpha = va[h]; seg = 8 + h; }
    float acc = bias;
    #pragma unroll 8
    for (int c = 0; c < C_; ++c) acc += wrow[c] * xs[c * F_ + f];
    float y = (acc >= 0.f) ? acc : alpha * acc;      // PReLU
    ys[idx] = y;
    atomicAdd(&segsum[seg], y);                      // ds_add_f32
    atomicAdd(&segss[seg], y * y);
  }
  __syncthreads();

  if (tid < 12) {
    float cnt = (tid < 8) ? (float)(HD_ * F_) : (float)(CVH_ * F_);
    float mu = segsum[tid] / cnt;
    float var = segss[tid] / cnt - mu * mu;
    smu[tid] = mu;
    srstd[tid] = rsqrtf(var + 1e-5f);
  }
  __syncthreads();

  for (int idx = tid; idx < 128 * F_; idx += 256) {
    int r = idx / F_, f = idx % F_;
    if (r < 32) {
      int h = r >> 3, o = r & 7, seg = h;
      float v = (ys[idx] - smu[seg]) * srstd[seg] * qg[r * F_ + f] + qbe[r * F_ + f];
      int n = h * B_ + b;
      Qh[((size_t)n * T_ + t) * DQKP + o * F_ + f] = (_Float16)(v * scale);
    } else if (r < 64) {
      int rr = r - 32, h = rr >> 3, o = rr & 7, seg = 4 + h;
      float v = (ys[idx] - smu[seg]) * srstd[seg] * kg[rr * F_ + f] + kbe[rr * F_ + f];
      int n = h * B_ + b;
      Kh[((size_t)n * T_ + t) * DQKP + o * F_ + f] = (_Float16)v;
    } else {
      int rr = r - 64, h = rr >> 4, o = rr & 15, seg = 8 + h;
      float v = (ys[idx] - smu[seg]) * srstd[seg] * vg[rr * F_ + f] + vbe[rr * F_ + f];
      int n = h * B_ + b;
      Vt[((size_t)n * DVP + o * F_ + f) * T_ + t] = (_Float16)v;   // transposed store
    }
  }
  // zero-pad Q/K columns d in [520,544)
  for (int p = tid; p < 2 * H_ * (DQKP - DQK); p += 256) {
    int per = H_ * (DQKP - DQK);
    int which = p / per, rem = p % per;
    int h = rem / (DQKP - DQK);
    int d = DQK + rem % (DQKP - DQK);
    int n = h * B_ + b;
    _Float16* dst = which ? Kh : Qh;
    dst[((size_t)n * T_ + t) * DQKP + d] = (_Float16)0.f;
  }
  // zero-pad Vt rows d in [1040,1088)
  for (int p = tid; p < H_ * (DVP - DV); p += 256) {
    int h = p / (DVP - DV);
    int d = DV + p % (DVP - DV);
    int n = h * B_ + b;
    Vt[((size_t)n * DVP + d) * T_ + t] = (_Float16)0.f;
  }
}

// ---------------------------------------------------------------------------
// Stage 2: S = (Q*scale) K^T  (f16 logits).  One 64x64 macro-tile per wave:
// 4x4 register blocking -> 4 A-frags + 4 B-frags feed 16 WMMAs per K-step
// (0.5 KB of loads per 16.4 KFLOP wmma instead of 2 KB).
// A-frag: lane(row = lane&15), K in {kb..kb+7} U {kb+16..kb+23}, kb=(lane>>4)*8
// B-frag: lane(col = lane&15), K in {kb..kb+7} U {kb+16..kb+23}, kb=(lane>>4)*8
// ---------------------------------------------------------------------------
__global__ __launch_bounds__(256) void qk_logits_kernel(
    const _Float16* __restrict__ Qh, const _Float16* __restrict__ Kh,
    _Float16* __restrict__ P)
{
  const int lane = threadIdx.x & 31;
  const int wave = threadIdx.x >> 5;
  const int macro = blockIdx.x * 8 + wave;   // 0..255 per n
  const int tm = macro >> 4, sm = macro & 15;
  const int n = blockIdx.y;
  const int mrow = lane & 15;
  const int kbA = (lane >> 4) * 8;
  const int kbB = (lane >> 4) * 8;
  const _Float16* qp = Qh + ((size_t)n * T_ + tm * 64 + mrow) * DQKP + kbA;
  const _Float16* kp = Kh + ((size_t)n * T_ + sm * 64 + mrow) * DQKP + kbB;

  v8f acc[4][4] = {};
  for (int k0 = 0; k0 < DQKP; k0 += 32) {
    __builtin_prefetch(qp + k0 + 128, 0, 1);   // global_prefetch_b8
    __builtin_prefetch(kp + k0 + 128, 0, 1);
    union { v16h v; v8h h[2]; } Af[4], Bf[4];
    #pragma unroll
    for (int i = 0; i < 4; ++i) {
      const _Float16* p0 = qp + (size_t)i * 16 * DQKP + k0;
      Af[i].h[0] = *(const v8h*)(p0);
      Af[i].h[1] = *(const v8h*)(p0 + 16);
    }
    #pragma unroll
    for (int j = 0; j < 4; ++j) {
      const _Float16* q0 = kp + (size_t)j * 16 * DQKP + k0;
      Bf[j].h[0] = *(const v8h*)(q0);
      Bf[j].h[1] = *(const v8h*)(q0 + 16);
    }
    #pragma unroll
    for (int i = 0; i < 4; ++i)
      #pragma unroll
      for (int j = 0; j < 4; ++j)
        acc[i][j] = __builtin_amdgcn_wmma_f32_16x16x32_f16(
            false, Af[i].v, false, Bf[j].v, (short)0, acc[i][j], false, false);
  }

  const int rbase = (lane >> 4) * 8;
  #pragma unroll
  for (int i = 0; i < 4; ++i) {
    #pragma unroll
    for (int j = 0; j < 4; ++j) {
      int scol = sm * 64 + j * 16 + (lane & 15);
      #pragma unroll
      for (int r = 0; r < 8; ++r) {
        int trow = tm * 64 + i * 16 + rbase + r;
        P[((size_t)n * T_ + trow) * T_ + scol] = (_Float16)acc[i][j][r];
      }
    }
  }
}

// ---------------------------------------------------------------------------
// Stage 3: row softmax over P (in place), one block per row.
// ---------------------------------------------------------------------------
__global__ __launch_bounds__(256) void softmax_kernel(_Float16* __restrict__ P)
{
  __shared__ float red[256];
  const int tid = threadIdx.x;
  _Float16* row = P + (size_t)blockIdx.x * T_;

  float v[4];
  float mx = -3.0e38f;
  #pragma unroll
  for (int i = 0; i < 4; ++i) { v[i] = (float)row[tid + i * 256]; mx = fmaxf(mx, v[i]); }
  red[tid] = mx; __syncthreads();
  for (int s = 128; s > 0; s >>= 1) { if (tid < s) red[tid] = fmaxf(red[tid], red[tid + s]); __syncthreads(); }
  mx = red[0]; __syncthreads();

  float sum = 0.f;
  #pragma unroll
  for (int i = 0; i < 4; ++i) { v[i] = __expf(v[i] - mx); sum += v[i]; }
  red[tid] = sum; __syncthreads();
  for (int s = 128; s > 0; s >>= 1) { if (tid < s) red[tid] += red[tid + s]; __syncthreads(); }
  float inv = 1.0f / red[0];
  #pragma unroll
  for (int i = 0; i < 4; ++i) row[tid + i * 256] = (_Float16)(v[i] * inv);
}

// ---------------------------------------------------------------------------
// Stage 4: A = P @ V  (K = T = 1024).  64x64 macro-tile per wave (4x4 blocking),
// f32 results scattered directly into the (B,C,T,F) layout.  N padded to 1088;
// sub-tiles with d >= 1040 are computed against zero pad and not stored.
// ---------------------------------------------------------------------------
__global__ __launch_bounds__(256) void pv_kernel(
    const _Float16* __restrict__ P, const _Float16* __restrict__ Vt,
    float* __restrict__ A)
{
  const int lane = threadIdx.x & 31;
  const int wave = threadIdx.x >> 5;
  const int macro = blockIdx.x * 8 + wave;    // 0..271 per n (16 t-macros x 17 d-macros)
  const int tm = macro / 17, dm = macro % 17;
  const int n = blockIdx.y;
  const int mrow = lane & 15;
  const int kbA = (lane >> 4) * 8;
  const int kbB = (lane >> 4) * 8;
  const _Float16* pp = P  + ((size_t)n * T_  + tm * 64 + mrow) * T_ + kbA;
  const _Float16* vp = Vt + ((size_t)n * DVP + dm * 64 + mrow) * T_ + kbB;

  v8f acc[4][4] = {};
  for (int k0 = 0; k0 < T_; k0 += 32) {
    __builtin_prefetch(pp + k0 + 128, 0, 1);
    __builtin_prefetch(vp + k0 + 128, 0, 1);
    union { v16h v; v8h h[2]; } Af[4], Bf[4];
    #pragma unroll
    for (int i = 0; i < 4; ++i) {
      const _Float16* p0 = pp + (size_t)i * 16 * T_ + k0;
      Af[i].h[0] = *(const v8h*)(p0);
      Af[i].h[1] = *(const v8h*)(p0 + 16);
    }
    #pragma unroll
    for (int j = 0; j < 4; ++j) {
      const _Float16* q0 = vp + (size_t)j * 16 * T_ + k0;
      Bf[j].h[0] = *(const v8h*)(q0);
      Bf[j].h[1] = *(const v8h*)(q0 + 16);
    }
    #pragma unroll
    for (int i = 0; i < 4; ++i)
      #pragma unroll
      for (int j = 0; j < 4; ++j)
        acc[i][j] = __builtin_amdgcn_wmma_f32_16x16x32_f16(
            false, Af[i].v, false, Bf[j].v, (short)0, acc[i][j], false, false);
  }

  const int h = n >> 2, b = n & 3;
  const int rbase = (lane >> 4) * 8;
  #pragma unroll
  for (int j = 0; j < 4; ++j) {
    int d = dm * 64 + j * 16 + (lane & 15);   // 16-aligned tiles: all-or-nothing vs 1040
    if (d < DV) {
      int cv = d / F_, f = d % F_;
      int c = h * CVH_ + cv;
      #pragma unroll
      for (int i = 0; i < 4; ++i) {
        #pragma unroll
        for (int r = 0; r < 8; ++r) {
          int trow = tm * 64 + i * 16 + rbase + r;
          A[(((size_t)b * C_ + c) * T_ + trow) * F_ + f] = acc[i][j][r];
        }
      }
    }
  }
}

// ---------------------------------------------------------------------------
// Stage 5: out = cfLN(PReLU(p_w @ A + p_b)) + x.  One block per (t, b).
// ---------------------------------------------------------------------------
__global__ __launch_bounds__(256) void out_proj_kernel(
    const float* __restrict__ A, const float* __restrict__ x,
    const float* __restrict__ pw, const float* __restrict__ pb,
    const float* __restrict__ pa, const float* __restrict__ pg,
    const float* __restrict__ pbe, float* __restrict__ out)
{
  __shared__ float as[C_ * F_];
  __shared__ float ys[C_ * F_];
  __shared__ float wps[C_ * C_];
  __shared__ float pbs[C_];
  __shared__ float ssum, sss, smu, srstd;
  const int tid = threadIdx.x;
  const int t = blockIdx.x, b = blockIdx.y;

  for (int i = tid; i < C_ * F_; i += 256) {
    int c = i / F_, f = i % F_;
    as[i] = A[(((size_t)b * C_ + c) * T_ + t) * F_ + f];
  }
  for (int i = tid; i < C_ * C_; i += 256) wps[i] = pw[i];
  if (tid < C_) pbs[tid] = pb[tid];
  if (tid == 0) { ssum = 0.f; sss = 0.f; }
  __syncthreads();

  const float alpha = pa[0];
  float psum = 0.f, pss = 0.f;
  for (int idx = tid; idx < C_ * F_; idx += 256) {
    int o = idx / F_, f = idx % F_;
    float acc = pbs[o];
    #pragma unroll 8
    for (int c = 0; c < C_; ++c) acc += wps[o * C_ + c] * as[c * F_ + f];
    float y = (acc >= 0.f) ? acc : alpha * acc;
    ys[idx] = y; psum += y; pss += y * y;
  }
  atomicAdd(&ssum, psum);
  atomicAdd(&sss, pss);
  __syncthreads();
  if (tid == 0) {
    float cnt = (float)(C_ * F_);
    float mu = ssum / cnt;
    float var = sss / cnt - mu * mu;
    smu = mu; srstd = rsqrtf(var + 1e-5f);
  }
  __syncthreads();

  for (int idx = tid; idx < C_ * F_; idx += 256) {
    int o = idx / F_, f = idx % F_;
    size_t g = (((size_t)b * C_ + o) * T_ + t) * F_ + f;
    out[g] = (ys[idx] - smu) * srstd * pg[idx] + pbe[idx] + x[g];
  }
}

// ---------------------------------------------------------------------------
extern "C" void kernel_launch(void* const* d_in, const int* in_sizes, int n_in,
                              void* d_out, int out_size, void* d_ws, size_t ws_size,
                              hipStream_t stream)
{
  const float* x   = (const float*)d_in[0];
  const float* qw  = (const float*)d_in[1];
  const float* qb  = (const float*)d_in[2];
  const float* qa  = (const float*)d_in[3];
  const float* qg  = (const float*)d_in[4];
  const float* qbe = (const float*)d_in[5];
  const float* kw  = (const float*)d_in[6];
  const float* kb  = (const float*)d_in[7];
  const float* ka  = (const float*)d_in[8];
  const float* kg  = (const float*)d_in[9];
  const float* kbe = (const float*)d_in[10];
  const float* vw  = (const float*)d_in[11];
  const float* vb  = (const float*)d_in[12];
  const float* va  = (const float*)d_in[13];
  const float* vg  = (const float*)d_in[14];
  const float* vbe = (const float*)d_in[15];
  const float* pw  = (const float*)d_in[16];
  const float* pb  = (const float*)d_in[17];
  const float* pa  = (const float*)d_in[18];
  const float* pg  = (const float*)d_in[19];
  const float* pbe = (const float*)d_in[20];
  float* out = (float*)d_out;

  // Workspace carve-up (~165 MB total)
  char* ws = (char*)d_ws;
  const size_t szQ = (size_t)NH * T_ * DQKP * sizeof(_Float16);   // 17,825,792
  const size_t szV = (size_t)NH * DVP * T_ * sizeof(_Float16);    // 35,651,584
  const size_t szP = (size_t)NH * T_ * T_ * sizeof(_Float16);     // 33,554,432
  _Float16* Qh = (_Float16*)(ws);
  _Float16* Kh = (_Float16*)(ws + szQ);
  _Float16* Vt = (_Float16*)(ws + 2 * szQ);
  _Float16* P  = (_Float16*)(ws + 2 * szQ + szV);
  float*    A  = (float*)   (ws + 2 * szQ + szV + szP);

  const float scale = 1.0f / sqrtf((float)(F_ * HD_));

  qkv_proj_kernel<<<dim3(T_, B_), 256, 0, stream>>>(
      x, qw, qb, qa, qg, qbe, kw, kb, ka, kg, kbe, vw, vb, va, vg, vbe,
      Qh, Kh, Vt, scale);

  // 256 macro-tiles (64x64) per n, 8 waves/block
  qk_logits_kernel<<<dim3(32, NH), 256, 0, stream>>>(Qh, Kh, P);

  softmax_kernel<<<dim3(NH * T_), 256, 0, stream>>>(P);

  // 16 x 17 = 272 macro-tiles per n, 8 waves/block
  pv_kernel<<<dim3(34, NH), 256, 0, stream>>>(P, Vt, A);

  out_proj_kernel<<<dim3(T_, B_), 256, 0, stream>>>(A, x, pw, pb, pa, pg, pbe, out);
}
